// SingleGCN_41394894798937
// MI455X (gfx1250) — matrix operand order repaired and checked by reading, not compiled, not run on hardware
//
#include <hip/hip_runtime.h>
#include <hip/hip_bf16.h>

typedef __attribute__((ext_vector_type(16))) _Float16 v16h;
typedef __attribute__((ext_vector_type(8)))  _Float16 v8h;
typedef __attribute__((ext_vector_type(8)))  float    v8f;

#define FEATS 128

// ---------------------------------------------------------------------------
// Kernel 1: zero accumulator + degree buffers
// ---------------------------------------------------------------------------
__global__ void gcn_zero_kernel(float* __restrict__ p, int n) {
    int i = blockIdx.x * blockDim.x + threadIdx.x;
    if (i < n) p[i] = 0.0f;
}

// ---------------------------------------------------------------------------
// Kernel 2: edge scatter. One wave32 per edge: lane L handles feats [4L,4L+4).
// hidden row read is one coalesced 512B burst; accumulation via HW f32 atomics
// that land in L2 (accum is 25.8MB << 192MB L2).
// ---------------------------------------------------------------------------
__global__ __launch_bounds__(256)
void gcn_scatter_kernel(const float* __restrict__ hidden,
                        const int*   __restrict__ src,
                        const int*   __restrict__ dst,
                        float* __restrict__ accum,
                        float* __restrict__ deg,
                        int n_edges) {
    int edge = (blockIdx.x * 256 + threadIdx.x) >> 5;
    int lane = threadIdx.x & 31;
    if (edge >= n_edges) return;

    int s = src[edge];
    int d = dst[edge];

    const float4 v = ((const float4*)(hidden + (size_t)s * FEATS))[lane];
    float* dp = accum + (size_t)d * FEATS + lane * 4;

    __hip_atomic_fetch_add(dp + 0, v.x, __ATOMIC_RELAXED, __HIP_MEMORY_SCOPE_AGENT);
    __hip_atomic_fetch_add(dp + 1, v.y, __ATOMIC_RELAXED, __HIP_MEMORY_SCOPE_AGENT);
    __hip_atomic_fetch_add(dp + 2, v.z, __ATOMIC_RELAXED, __HIP_MEMORY_SCOPE_AGENT);
    __hip_atomic_fetch_add(dp + 3, v.w, __ATOMIC_RELAXED, __HIP_MEMORY_SCOPE_AGENT);
    if (lane == 0)
        __hip_atomic_fetch_add(deg + d, 1.0f, __ATOMIC_RELAXED, __HIP_MEMORY_SCOPE_AGENT);
}

// ---------------------------------------------------------------------------
// Kernel 3: convert W (f32, [K][N] row-major) -> Wt (f16, [N][K] transposed)
// so that WMMA B-fragments become contiguous 16-half runs per lane.
// ---------------------------------------------------------------------------
__global__ void gcn_wcvt_kernel(const float* __restrict__ W,
                                _Float16* __restrict__ Wt) {
    int i = blockIdx.x * 256 + threadIdx.x;   // 0 .. 16383
    int n = i >> 7;
    int k = i & 127;
    Wt[n * FEATS + k] = (_Float16)W[k * FEATS + n];
}

// ---------------------------------------------------------------------------
// Kernel 4: mean + GEMM + bias + ReLU via v_wmma_f32_16x16x32_f16.
// 256 threads = 8 waves; each wave owns a 16-node tile x 128 output feats:
//   8 accumulator tiles (16x16 f32), K=128 in 4 steps of 32 -> 32 WMMAs/wave.
// A fragment (16x32 f16, contiguous K, same order as B): lane half h = lane>>4, row r = lane&15,
//   pair j (VGPR j): K = k0 + 16h + 2j.
// B fragment (32x16 f16): lane holds column N = r, K = k0 + e + 16h (contiguous
//   in Wt[N][K] -> single 32B LDS read).
// C/D tile: VGPR i -> M = i + 8h, N = r.
// ---------------------------------------------------------------------------
__global__ __launch_bounds__(256)
void gcn_gemm_kernel(const float*    __restrict__ accum,
                     const float*    __restrict__ deg,
                     const _Float16* __restrict__ Wt,
                     const float*    __restrict__ bias,
                     float*          __restrict__ out,
                     int n_nodes) {
    __shared__ v16h Wl[FEATS * FEATS / 16];   // 32 KB, layout Wt[N][K] in halves

    const int tid = threadIdx.x;

    // Stage f16 weights to LDS (1024 x 32B, 4 vectors per thread)
    {
        const v16h* g = (const v16h*)Wt;
        for (int i = tid; i < FEATS * FEATS / 16; i += 256) Wl[i] = g[i];
    }
    __syncthreads();

    const int wave = tid >> 5;
    const int lane = tid & 31;
    const int r    = lane & 15;
    const int h    = lane >> 4;

    const int tile  = blockIdx.x * 8 + wave;       // 16-node tile index
    const int node0 = tile * 16;
    if (node0 >= n_nodes) return;                  // wave-uniform exit (after barrier)

    int row = node0 + r;
    if (row >= n_nodes) row = n_nodes - 1;         // safety clamp (50000 % 16 == 0)

    const float dv   = deg[row];
    const float invd = 1.0f / fmaxf(dv, 1.0f);
    const float* arow = accum + (size_t)row * FEATS;

    v8f acc[8] = {};

    #pragma unroll
    for (int k0 = 0; k0 < FEATS; k0 += 32) {
        // Build A fragment with fused mean (accum * invd), f32 -> f16
        v16h a;
        #pragma unroll
        for (int j = 0; j < 8; ++j) {
            const int k = k0 + 16 * h + 2 * j;
            const float2 f = *(const float2*)(arow + k);
            a[2 * j]     = (_Float16)(f.x * invd);
            a[2 * j + 1] = (_Float16)(f.y * invd);
        }
        #pragma unroll
        for (int nt = 0; nt < 8; ++nt) {
            const int n = nt * 16 + r;
            // 16 contiguous halves: Wl[(n*128 + k0 + 16h) / 16]
            const v16h bf = Wl[n * 8 + (k0 >> 4) + h];
            acc[nt] = __builtin_amdgcn_wmma_f32_16x16x32_f16(
                false, a, false, bf, (short)0, acc[nt], false, false);
        }
    }

    // Epilogue: + bias, ReLU, store (C/D layout: VGPR i -> M = i + 8h, N = r)
    #pragma unroll
    for (int nt = 0; nt < 8; ++nt) {
        const int n  = nt * 16 + r;
        const float bb = bias[n];
        #pragma unroll
        for (int i = 0; i < 8; ++i) {
            const int m = node0 + i + 8 * h;
            if (m < n_nodes) {
                const float v = acc[nt][i] + bb;
                out[(size_t)m * FEATS + n] = fmaxf(v, 0.0f);
            }
        }
    }
}

// ---------------------------------------------------------------------------
// Launch
// ---------------------------------------------------------------------------
extern "C" void kernel_launch(void* const* d_in, const int* in_sizes, int n_in,
                              void* d_out, int out_size, void* d_ws, size_t ws_size,
                              hipStream_t stream) {
    const float* hidden = (const float*)d_in[0];
    const float* W      = (const float*)d_in[1];
    const float* bias   = (const float*)d_in[2];
    const int*   src    = (const int*)d_in[3];
    const int*   dst    = (const int*)d_in[4];
    float*       out    = (float*)d_out;

    const int n_nodes = in_sizes[0] / FEATS;   // 50000
    const int n_edges = in_sizes[3];           // 800000

    // Workspace layout: [ accum f32 N*128 | deg f32 N | Wt f16 128*128 ]
    float*    accum = (float*)d_ws;
    float*    deg   = accum + (size_t)n_nodes * FEATS;
    _Float16* Wt    = (_Float16*)(deg + n_nodes);

    // 1) zero accum + deg (contiguous in ws)
    {
        const int n = n_nodes * FEATS + n_nodes;
        gcn_zero_kernel<<<(n + 255) / 256, 256, 0, stream>>>(accum, n);
    }

    // 2) edge scatter: one wave per edge -> 8 edges per 256-thread block
    {
        const int blocks = (n_edges + 7) / 8;
        gcn_scatter_kernel<<<blocks, 256, 0, stream>>>(hidden, src, dst,
                                                       accum, deg, n_edges);
    }

    // 3) W -> f16 transposed
    gcn_wcvt_kernel<<<(FEATS * FEATS) / 256, 256, 0, stream>>>(W, Wt);

    // 4) mean + WMMA GEMM + bias + ReLU
    {
        const int tiles  = (n_nodes + 15) / 16;   // 3125
        const int blocks = (tiles + 7) / 8;       // 391
        gcn_gemm_kernel<<<blocks, 256, 0, stream>>>(accum, deg, Wt, bias,
                                                    out, n_nodes);
    }
}
